// snn_2_47725676593666
// MI455X (gfx1250) — compile-verified, matrix operand order checked by reading, not run on hardware
//
#include <hip/hip_runtime.h>
#include <hip/hip_bf16.h>

// ---------------- problem constants (from reference) ----------------
#define BB      64
#define CC      3
#define N_WIN   232
#define HH      64
#define WW      64
#define TSTEPS  8
#define CHUNK   29          // N_WIN / TSTEPS
#define KD      4
#define KH      2
#define KW      2
#define DP      13          // (CHUNK-KD)/2 + 1
#define HP      32
#define WP      32
#define IN_DIM  1024        // HP*WP
#define H0      512
#define H1      512
#define H2      256
#define CD      39          // CC*DP
#define FEAT    9984        // CD*H2
#define NCLS    4
#define BETA    0.9f
#define THR     0.5f

#define MROWS   (BB * CD)   // 2496
#define M_PAD   2560        // 10 * 256, full tiles for WMMA (no edge predication)

typedef _Float16 v16h __attribute__((ext_vector_type(16)));
typedef _Float16 v8h  __attribute__((ext_vector_type(8)));
typedef float    v8f  __attribute__((ext_vector_type(8)));

// ---------------- workspace layout (bytes) ----------------
#define OFF_XT    0ull                                   // 8 * M_PAD * 1024 f16
#define SZ_XT     ((size_t)TSTEPS * M_PAD * IN_DIM * 2)
#define OFF_W1    (OFF_XT + SZ_XT)                       // 512x1024 f16
#define SZ_W1     ((size_t)H0 * IN_DIM * 2)
#define OFF_W2    (OFF_W1 + SZ_W1)                       // 512x512 f16
#define SZ_W2     ((size_t)H1 * H0 * 2)
#define OFF_W3    (OFF_W2 + SZ_W2)                       // 256x512 f16
#define SZ_W3     ((size_t)H2 * H1 * 2)
#define OFF_S1    (OFF_W3 + SZ_W3)
#define SZ_S1     ((size_t)M_PAD * H0 * 2)
#define OFF_S2    (OFF_S1 + SZ_S1)
#define SZ_S2     ((size_t)M_PAD * H1 * 2)
#define OFF_S3    (OFF_S2 + SZ_S2)
#define SZ_S3     ((size_t)M_PAD * H2 * 2)
#define OFF_MIN   (OFF_S3 + SZ_S3)
#define SZ_MIN    ((size_t)M_PAD * H0 * 4)
#define OFF_MH1   (OFF_MIN + SZ_MIN)
#define SZ_MH1    ((size_t)M_PAD * H1 * 4)
#define OFF_MH2   (OFF_MH1 + SZ_MH1)
#define SZ_MH2    ((size_t)M_PAD * H2 * 4)
#define OFF_MOUT  (OFF_MH2 + SZ_MH2)
#define SZ_MOUT   ((size_t)BB * NCLS * 4)

// ---------------- gfx1250 async memory->LDS (16B per lane, ASYNCcnt-tracked) ----------------
// LDS aperture maps flat addr[31:0] directly to the wave-relative LDS address,
// so the low 32 bits of a __shared__ pointer are the VDST LDS address.
__device__ __forceinline__ void async_copy_16B_to_lds(const _Float16* gsrc, _Float16* ldst) {
    unsigned loff = (unsigned)(size_t)ldst;
    asm volatile("global_load_async_to_lds_b128 %0, %1, off"
                 :: "v"(loff), "v"(gsrc) : "memory");
}
__device__ __forceinline__ void wait_asynccnt0() {
    asm volatile("s_wait_asynccnt 0" ::: "memory");
}

// ---------------- f32 -> f16 weight conversion ----------------
__global__ void f32_to_f16(const float* __restrict__ in, _Float16* __restrict__ out, int n) {
    int i = blockIdx.x * 256 + threadIdx.x;
    if (i < n) out[i] = (_Float16)in[i];
}

// ---------------- fused avg-pool3d + cast to f16 ----------------
// output layout: xt[t][m = b*CD + c*DP + d][hp*WP + wp], row stride IN_DIM, t stride M_PAD*IN_DIM
__global__ __launch_bounds__(256) void pool_f16(const float* __restrict__ x,
                                                _Float16* __restrict__ xt) {
    size_t i = (size_t)blockIdx.x * 256 + threadIdx.x;
    const size_t total = (size_t)TSTEPS * BB * CC * DP * HP * WP;
    if (i >= total) return;
    int wp = i % WP;  size_t r = i / WP;
    int hp = r % HP;  r /= HP;
    int d  = r % DP;  r /= DP;
    int c  = r % CC;  r /= CC;
    int b  = r % BB;  int t = (int)(r / BB);

    const int n0 = t * CHUNK + d * 2;        // depth window start along N_WIN
    const int h0 = hp * 2, w0 = wp * 2;
    const size_t basebc = ((size_t)b * CC + c) * N_WIN;
    float acc = 0.0f;
    #pragma unroll
    for (int kd = 0; kd < KD; ++kd) {
        const float* p = x + ((basebc + n0 + kd) * HH + h0) * WW + w0;
        acc += p[0] + p[1] + p[WW] + p[WW + 1];
    }
    acc *= (1.0f / (KD * KH * KW));

    const size_t m = (size_t)b * CD + c * DP + d;
    xt[((size_t)t * M_PAD + m) * IN_DIM + hp * WP + wp] = (_Float16)acc;
}

// ---------------- fused WMMA GEMM + LIF ----------------
// C[m][n] = sum_k A[m][k] * W[n][k] + bias[n]; then LIF update of mem, spike -> f16
// Block: 256 threads = 8 waves; tile 256 (rows) x 64 (cols); wave w owns rows 32w..32w+31
// (2 m-tiles x 4 n-tiles = 8 WMMAs per K-step, pairs sharing each B fragment).
// Weight tile double-buffered in LDS and staged with GLOBAL_LOAD_ASYNC_TO_LDS_B128:
// no VGPR round trip, no loadcnt stall; one s_wait_asynccnt + barrier per K-iteration.
__global__ __launch_bounds__(256) void gemm_lif(const _Float16* __restrict__ A,   // M_PAD x K
                                                const _Float16* __restrict__ W,   // N x K
                                                const float*    __restrict__ bias,// N
                                                float*          __restrict__ mem, // M_PAD x N
                                                _Float16*       __restrict__ spk, // M_PAD x N
                                                int N, int K) {
    __shared__ alignas(16) _Float16 Bs[2][64][40];   // 2 x (64 weight rows x 32 halfs (+pad))

    const int tid   = threadIdx.x;
    const int wave  = tid >> 5;
    const int lane  = tid & 31;
    const int l15   = lane & 15;
    const int khalf = (lane < 16) ? 0 : 8;        // ISA 16-bit A-fragment K striping
    const int row0  = blockIdx.x * 256 + wave * 32;
    const int nbase = blockIdx.y * 64;

    v8f acc[2][4];
    #pragma unroll
    for (int mt = 0; mt < 2; ++mt)
        #pragma unroll
        for (int nt = 0; nt < 4; ++nt) acc[mt][nt] = v8f{};

    const int brow = tid >> 2;                    // 0..63  (B-tile stage row)
    const int bcol = (tid & 3) * 8;               // 0,8,16,24 halfs
    const _Float16* wrow = W + (size_t)(nbase + brow) * K + bcol;
    const _Float16* arow0 = A + (size_t)(row0 + l15) * K + khalf;
    const _Float16* arow1 = arow0 + (size_t)16 * K;

    // preload weight tile 0 via async DMA, then make it visible block-wide
    async_copy_16B_to_lds(wrow, &Bs[0][brow][bcol]);
    wait_asynccnt0();
    __syncthreads();

    int buf = 0;
    for (int k0 = 0; k0 < K; k0 += 32) {
        // DMA next weight tile into the other buffer while consuming this one
        if (k0 + 32 < K) {
            async_copy_16B_to_lds(wrow + k0 + 32, &Bs[buf ^ 1][brow][bcol]);
            __builtin_prefetch(arow0 + k0 + 32, 0, 3);
            __builtin_prefetch(arow1 + k0 + 32, 0, 3);
        }

        // A fragments (2 m-tiles): halfs [0..7]=K khalf..+7, [8..15]=K khalf+16..+23
        v16h af0, af1;
        ((v8h*)&af0)[0] = *(const v8h*)(arow0 + k0);
        ((v8h*)&af0)[1] = *(const v8h*)(arow0 + k0 + 16);
        ((v8h*)&af1)[0] = *(const v8h*)(arow1 + k0);
        ((v8h*)&af1)[1] = *(const v8h*)(arow1 + k0 + 16);

        // 4 B fragments, each feeding 2 back-to-back WMMAs (m-tile pair)
        v16h bf[4];
        #pragma unroll
        for (int nt = 0; nt < 4; ++nt) {
            const _Float16* bp = &Bs[buf][nt * 16 + l15][khalf];
            ((v8h*)&bf[nt])[0] = *(const v8h*)(bp);
            ((v8h*)&bf[nt])[1] = *(const v8h*)(bp + 16);
        }

        #pragma unroll
        for (int nt = 0; nt < 4; ++nt) {
            acc[0][nt] = __builtin_amdgcn_wmma_f32_16x16x32_f16(
                false, af0, false, bf[nt], (short)0, acc[0][nt], false, false);
            acc[1][nt] = __builtin_amdgcn_wmma_f32_16x16x32_f16(
                false, af1, false, bf[nt], (short)0, acc[1][nt], false, false);
        }

        wait_asynccnt0();  // next tile's DMA landed in LDS
        __syncthreads();   // all waves done reading Bs[buf] / writing Bs[buf^1]
        buf ^= 1;
    }

    // fused LIF epilogue: D layout — VGPR r, lanes 0-15: M=r, lanes 16-31: M=r+8; N = lane&15
    const int mofs = (lane >> 4) << 3;            // 0 or 8
    #pragma unroll
    for (int mt = 0; mt < 2; ++mt) {
        #pragma unroll
        for (int nt = 0; nt < 4; ++nt) {
            const int n = nbase + nt * 16 + l15;
            const float bv = bias[n];
            #pragma unroll
            for (int r = 0; r < 8; ++r) {
                const int m = row0 + mt * 16 + r + mofs;
                const size_t idx = (size_t)m * N + n;
                const float cur = acc[mt][nt][r] + bv;
                const float mo  = mem[idx];
                const float rst = (mo > THR) ? THR : 0.0f;    // reset-by-subtraction, detached
                const float mn  = BETA * mo + cur - rst;
                mem[idx] = mn;
                spk[idx] = (mn > THR) ? (_Float16)1.0f : (_Float16)0.0f;
            }
        }
    }
}

// ---------------- output layer: (64 x 9984) @ (9984 x 4) + LIF(thr=1) ----------------
__global__ __launch_bounds__(256) void out_lif(const _Float16* __restrict__ s3,   // M_PAD x H2
                                               const float* __restrict__ Wout,    // 4 x FEAT
                                               const float* __restrict__ bout,    // 4
                                               float* __restrict__ mout,          // 64 x 4
                                               float* __restrict__ out, int t) {
    const int b   = blockIdx.x;
    const int tid = threadIdx.x;
    // rows b*CD .. b*CD+CD-1 of s3 are contiguous -> flat FEAT vector
    const _Float16* sp = s3 + (size_t)b * CD * H2;

    float a0 = 0.f, a1 = 0.f, a2 = 0.f, a3 = 0.f;
    for (int k = tid; k < FEAT; k += 256) {
        const float sv = (float)sp[k];
        a0 += sv * Wout[0 * FEAT + k];
        a1 += sv * Wout[1 * FEAT + k];
        a2 += sv * Wout[2 * FEAT + k];
        a3 += sv * Wout[3 * FEAT + k];
    }
    __shared__ float red[4][256];
    red[0][tid] = a0; red[1][tid] = a1; red[2][tid] = a2; red[3][tid] = a3;
    __syncthreads();
    for (int s = 128; s > 0; s >>= 1) {
        if (tid < s) {
            red[0][tid] += red[0][tid + s];
            red[1][tid] += red[1][tid + s];
            red[2][tid] += red[2][tid + s];
            red[3][tid] += red[3][tid + s];
        }
        __syncthreads();
    }
    if (tid < NCLS) {
        const float cur = red[tid][0] + bout[tid];
        const float mo  = mout[b * NCLS + tid];
        const float rst = (mo > 1.0f) ? 1.0f : 0.0f;      // output LIF threshold = 1.0
        const float mn  = BETA * mo + cur - rst;
        mout[b * NCLS + tid] = mn;
        out[((size_t)t * BB + b) * NCLS + tid] = (mn > 1.0f) ? 1.0f : 0.0f;
    }
}

extern "C" void kernel_launch(void* const* d_in, const int* in_sizes, int n_in,
                              void* d_out, int out_size, void* d_ws, size_t ws_size,
                              hipStream_t stream) {
    const float* x     = (const float*)d_in[0];
    const float* W_in  = (const float*)d_in[1];
    const float* b_in  = (const float*)d_in[2];
    const float* W_h1  = (const float*)d_in[3];
    const float* b_h1  = (const float*)d_in[4];
    const float* W_h2  = (const float*)d_in[5];
    const float* b_h2  = (const float*)d_in[6];
    const float* W_out = (const float*)d_in[7];
    const float* b_out = (const float*)d_in[8];

    char* ws = (char*)d_ws;
    _Float16* xt  = (_Float16*)(ws + OFF_XT);
    _Float16* w1h = (_Float16*)(ws + OFF_W1);
    _Float16* w2h = (_Float16*)(ws + OFF_W2);
    _Float16* w3h = (_Float16*)(ws + OFF_W3);
    _Float16* s1  = (_Float16*)(ws + OFF_S1);
    _Float16* s2  = (_Float16*)(ws + OFF_S2);
    _Float16* s3  = (_Float16*)(ws + OFF_S3);
    float*    mIn = (float*)(ws + OFF_MIN);
    float*    mH1 = (float*)(ws + OFF_MH1);
    float*    mH2 = (float*)(ws + OFF_MH2);
    float*    mOt = (float*)(ws + OFF_MOUT);

    // deterministic per-call state: zero all membranes (graph-capturable memset nodes)
    hipMemsetAsync(mIn, 0, SZ_MIN + SZ_MH1 + SZ_MH2 + SZ_MOUT, stream);

    // weight conversion f32 -> f16 (spikes are exactly representable; WMMA accumulates f32)
    f32_to_f16<<<(H0 * IN_DIM + 255) / 256, 256, 0, stream>>>(W_in, w1h, H0 * IN_DIM);
    f32_to_f16<<<(H1 * H0 + 255) / 256, 256, 0, stream>>>(W_h1, w2h, H1 * H0);
    f32_to_f16<<<(H2 * H1 + 255) / 256, 256, 0, stream>>>(W_h2, w3h, H2 * H1);

    // fused pooling: single pass over the 730 MB input (the bandwidth floor)
    {
        const size_t total = (size_t)TSTEPS * BB * CC * DP * HP * WP;
        pool_f16<<<(unsigned)((total + 255) / 256), 256, 0, stream>>>(x, xt);
    }

    // sequential timestep scan (stream order carries the membrane dependencies)
    for (int t = 0; t < TSTEPS; ++t) {
        const _Float16* At = xt + (size_t)t * M_PAD * IN_DIM;
        gemm_lif<<<dim3(M_PAD / 256, H0 / 64), 256, 0, stream>>>(At, w1h, b_in, mIn, s1, H0, IN_DIM);
        gemm_lif<<<dim3(M_PAD / 256, H1 / 64), 256, 0, stream>>>(s1, w2h, b_h1, mH1, s2, H1, H0);
        gemm_lif<<<dim3(M_PAD / 256, H2 / 64), 256, 0, stream>>>(s2, w3h, b_h2, mH2, s3, H2, H1);
        out_lif<<<BB, 256, 0, stream>>>(s3, W_out, b_out, mOt, (float*)d_out, t);
    }
}
